// TransformerAttention_57123065037443
// MI455X (gfx1250) — matrix operand order repaired and checked by reading, not compiled, not run on hardware
//
#include <hip/hip_runtime.h>
#include <hip/hip_bf16.h>

typedef __attribute__((ext_vector_type(16))) __bf16 v16bf;
typedef __attribute__((ext_vector_type(8)))  __bf16 v8bf;
typedef __attribute__((ext_vector_type(8)))  float  v8f;

#define BATCH 4
#define SEQ   2048
#define EMB   512
#define NHEAD 8
#define DK    64
#define DV    64
#define QKVC  1536            // H*(2*NK+NV)
#define ROWS  (BATCH*SEQ)     // 8192

static __device__ __forceinline__ unsigned short f2bf(float f) {
  return __builtin_bit_cast(unsigned short, (__bf16)f);
}
static __device__ __forceinline__ v16bf cat8(v8bf lo, v8bf hi) {
  v16bf r;
#pragma unroll
  for (int i = 0; i < 8; ++i) { r[i] = lo[i]; r[i + 8] = hi[i]; }
  return r;
}

// CDNA5 async global->LDS copy: 16 bytes per lane, tracked by ASYNCcnt.
// ldsByteOff is a literal LDS byte address (dynamic LDS starts at offset 0).
static __device__ __forceinline__ void async16(const void* g, unsigned ldsByteOff) {
  asm volatile("global_load_async_to_lds_b128 %0, %1, off"
               :: "v"(ldsByteOff), "v"(g) : "memory");
}
static __device__ __forceinline__ void wait_async0() {
  asm volatile("s_wait_asynccnt 0x0" ::: "memory");
}

// ---------------- conversion / transpose kernels ----------------
__global__ void k_cvt_x(const float* __restrict__ x, unsigned short* __restrict__ xb) {
  int i = (blockIdx.x * blockDim.x + threadIdx.x) * 4;
  float4 v = *(const float4*)(x + i);
  xb[i + 0] = f2bf(v.x); xb[i + 1] = f2bf(v.y);
  xb[i + 2] = f2bf(v.z); xb[i + 3] = f2bf(v.w);
}
__global__ void k_cvt_wqkvT(const float* __restrict__ w, unsigned short* __restrict__ wt) {
  int i = blockIdx.x * blockDim.x + threadIdx.x;   // over E*QKVC, k-major
  int k = i / QKVC, c = i % QKVC;
  wt[c * EMB + k] = f2bf(w[i]);
}
__global__ void k_cvt_woutT(const float* __restrict__ w, unsigned short* __restrict__ wt) {
  int i = blockIdx.x * blockDim.x + threadIdx.x;   // over (H*NV)*E, hv-major
  int hv = i / EMB, e = i % EMB;
  wt[e * EMB + hv] = f2bf(w[i]);
}

// ---------------- wave-level 16x16 bf16 WMMA GEMM tile ----------------
static __device__ __forceinline__ v8f wave_gemm_bf16(
    const unsigned short* __restrict__ A, int lda,
    const unsigned short* __restrict__ BT, int ldb, int kdim, int lane)
{
  const int m  = lane & 15;
  const int hi = lane >> 4;
  const unsigned short* arow = A + m * lda;
  const unsigned short* brow = BT + m * ldb + hi * 16;
  v8f acc = {};
  for (int k0 = 0; k0 < kdim; k0 += 32) {
    v8bf alo = *(const v8bf*)(arow + k0 + hi * 16);
    v8bf ahi = *(const v8bf*)(arow + k0 + hi * 16 + 8);
    v16bf a  = cat8(alo, ahi);
    v16bf b  = *(const v16bf*)(brow + k0);
    acc = __builtin_amdgcn_wmma_f32_16x16x32_bf16(false, a, false, b,
                                                  (short)0, acc, false, false);
  }
  return acc;
}

// ---------------- QKV projection (Q pre-scaled by 1/sqrt(dk)) ----------------
__global__ void __launch_bounds__(128) k_qkv(
    const unsigned short* __restrict__ Xb,
    const unsigned short* __restrict__ WqkvT,
    const float* __restrict__ bqkv,
    unsigned short* __restrict__ Q,
    unsigned short* __restrict__ K,
    unsigned short* __restrict__ Vt)
{
  const int lane = threadIdx.x & 31;
  const int wid  = blockIdx.x * 4 + (threadIdx.x >> 5);
  const int mt = wid / (QKVC / 16);
  const int nt = wid % (QKVC / 16);
  const int row0 = mt * 16, col0 = nt * 16;

  v8f acc = wave_gemm_bf16(Xb + (size_t)row0 * EMB, EMB,
                           WqkvT + (size_t)col0 * EMB, EMB, EMB, lane);

  const int hi = lane >> 4, n = lane & 15;
  const int gc = col0 + n;
  const int h = gc / 192, rr = gc % 192;
  const float bias = bqkv[gc];
#pragma unroll
  for (int r = 0; r < 8; ++r) {
    int gm = row0 + r + 8 * hi;
    int b = gm >> 11, s = gm & (SEQ - 1);
    float val = acc[r] + bias;
    if (rr < 64)
      Q[(((size_t)(b * NHEAD + h) * SEQ) + s) * DK + rr] = f2bf(val * 0.125f);
    else if (rr < 128)
      K[(((size_t)(b * NHEAD + h) * SEQ) + s) * DK + (rr - 64)] = f2bf(val);
    else
      Vt[(((size_t)(b * NHEAD + h) * DV) + (rr - 128)) * SEQ + s] = f2bf(val);
  }
}

// ---------------- flash attention ----------------
// Block = 4 waves, same (b,h), 4 consecutive 16-row q tiles.
// K/V chunks double-buffered in LDS via async global->LDS copies.
// LDS layout (ushort offsets), padded strides for conflict-free b128 reads:
//   K tiles : 2 x [32 rows x 72]   (64 data + 8 pad)   ->  9216 B
//   V tiles : 2 x [64 rows x 40]   (32 data + 8 pad)   -> 10240 B
//   P stage : 4 waves x [16 rows x 40]                 ->  5120 B
#define KOFF(bb) ((bb) * 2304)
#define VOFF(bb) (4608 + (bb) * 2560)
#define POFF0    9728
#define ATTN_LDS 24576

__global__ void __launch_bounds__(128) k_attn(
    const unsigned short* __restrict__ Q,
    const unsigned short* __restrict__ K,
    const unsigned short* __restrict__ Vt,
    const unsigned char* __restrict__ mask,
    unsigned short* __restrict__ Ctx)
{
  extern __shared__ unsigned short smem[];
  const int tid  = threadIdx.x;
  const int lane = tid & 31;
  const int w    = tid >> 5;
  const int wid  = blockIdx.x * 4 + w;
  const int qt = wid & 127;
  const int h  = (wid >> 7) & 7;
  const int b  = wid >> 10;
  const int m  = lane & 15, hi = lane >> 4;

  const unsigned short* Qp = Q  + (((size_t)(b * NHEAD + h) * SEQ) + qt * 16) * DK;
  const unsigned short* Kp = K  + ((size_t)(b * NHEAD + h) * SEQ) * DK;
  const unsigned short* Vp = Vt + ((size_t)(b * NHEAD + h) * DV) * SEQ;

  // async-copy slice owned by this thread
  const int kt = tid >> 3, ko = (tid & 7) * 8;   // K rows 0..15 (+16), 8x16B per row
  const int vr = tid >> 2, vo = (tid & 3) * 8;   // V rows 0..31 (+32), 4x16B per row
  auto issue = [&](int j, int bb) {
    const unsigned short* gk = Kp + (size_t)(j * 32 + kt) * DK + ko;
    async16(gk,           (unsigned)((KOFF(bb) + kt * 72 + ko) * 2));
    async16(gk + 16 * DK, (unsigned)((KOFF(bb) + (kt + 16) * 72 + ko) * 2));
    const unsigned short* gv = Vp + (size_t)vr * SEQ + j * 32 + vo;
    async16(gv,                  (unsigned)((VOFF(bb) + vr * 40 + vo) * 2));
    async16(gv + (size_t)32*SEQ, (unsigned)((VOFF(bb) + (vr + 32) * 40 + vo) * 2));
  };

  // Q A-fragments (already pre-scaled by 1/8)
  const unsigned short* qrow = Qp + m * DK;
  v16bf qa0 = cat8(*(const v8bf*)(qrow + hi * 16),      *(const v8bf*)(qrow + hi * 16 + 8));
  v16bf qa1 = cat8(*(const v8bf*)(qrow + 32 + hi * 16), *(const v8bf*)(qrow + 40 + hi * 16));

  v16bf ones;
#pragma unroll
  for (int i = 0; i < 16; ++i) ones[i] = (__bf16)1.0f;

  v8f o0 = {}, o1 = {}, o2 = {}, o3 = {}, o4 = {};   // o4 = row-sum (ones column trick)
  float mrow[8];
#pragma unroll
  for (int r = 0; r < 8; ++r) mrow[r] = -__builtin_inff();

  issue(0, 0);
  for (int j = 0; j < SEQ / 32; ++j) {
    const int bb = j & 1;
    wait_async0();          // this wave's slice of chunk j landed
    __syncthreads();        // all waves' slices landed; chunk j-1 fully consumed
    if (j + 1 < SEQ / 32) issue(j + 1, bb ^ 1);

    const unsigned short* kb = smem + KOFF(bb);
    const unsigned short* vb = smem + VOFF(bb);

    // ---- S = Q * K^T, two 16-col tiles ----
    v16bf b0 = *(const v16bf*)(kb + m * 72 + hi * 16);
    v16bf b1 = *(const v16bf*)(kb + m * 72 + 32 + hi * 16);
    v16bf c0 = *(const v16bf*)(kb + (16 + m) * 72 + hi * 16);
    v16bf c1 = *(const v16bf*)(kb + (16 + m) * 72 + 32 + hi * 16);
    v8f s0 = {}, s1 = {};
    s0 = __builtin_amdgcn_wmma_f32_16x16x32_bf16(false, qa0, false, b0, (short)0, s0, false, false);
    s0 = __builtin_amdgcn_wmma_f32_16x16x32_bf16(false, qa1, false, b1, (short)0, s0, false, false);
    s1 = __builtin_amdgcn_wmma_f32_16x16x32_bf16(false, qa0, false, c0, (short)0, s1, false, false);
    s1 = __builtin_amdgcn_wmma_f32_16x16x32_bf16(false, qa1, false, c1, (short)0, s1, false, false);

    // ---- mask ----
    const int kc = j * 32 + m;
#pragma unroll
    for (int r = 0; r < 8; ++r) {
      const int q = qt * 16 + r + 8 * hi;
      const size_t mb = (((size_t)b * SEQ + q) * NHEAD + h) * SEQ;
      if (!mask[mb + kc])      s0[r] = -__builtin_inff();
      if (!mask[mb + kc + 16]) s1[r] = -__builtin_inff();
      if (r == 0 && j + 1 < SEQ / 32)
        __builtin_prefetch(mask + mb + kc + 32, 0, 0);   // global_prefetch_b8
    }

    // ---- online softmax: max-reduce across the 16-lane row halves ----
    float fac[8];
#pragma unroll
    for (int r = 0; r < 8; ++r) {
      float t = fmaxf(s0[r], s1[r]);
#pragma unroll
      for (int xm = 1; xm < 16; xm <<= 1) t = fmaxf(t, __shfl_xor(t, xm, 32));
      float nm = fmaxf(mrow[r], t);
      fac[r] = __expf(mrow[r] - nm);
      s0[r] = __expf(s0[r] - nm);
      s1[r] = __expf(s1[r] - nm);
      mrow[r] = nm;
    }
#pragma unroll
    for (int r = 0; r < 8; ++r) {
      o0[r] *= fac[r]; o1[r] *= fac[r]; o2[r] *= fac[r]; o3[r] *= fac[r]; o4[r] *= fac[r];
    }

    // ---- P: D-layout -> A-layout via per-wave LDS staging (stride 40) ----
    unsigned short* ps = smem + POFF0 + w * 640;
#pragma unroll
    for (int r = 0; r < 8; ++r) {
      ps[(r + 8 * hi) * 40 + m]      = f2bf(s0[r]);
      ps[(r + 8 * hi) * 40 + 16 + m] = f2bf(s1[r]);
    }
    asm volatile("" ::: "memory");   // LDS is in-order within a wave
    v16bf pa = cat8(*(const v8bf*)(ps + m * 40 + hi * 16),
                    *(const v8bf*)(ps + m * 40 + 8 + hi * 16));
    asm volatile("" ::: "memory");

    // ---- O += P * V ; l += P * 1 ----
    v16bf v0 = *(const v16bf*)(vb + m * 40 + hi * 16);
    v16bf v1 = *(const v16bf*)(vb + (16 + m) * 40 + hi * 16);
    v16bf v2 = *(const v16bf*)(vb + (32 + m) * 40 + hi * 16);
    v16bf v3 = *(const v16bf*)(vb + (48 + m) * 40 + hi * 16);
    o0 = __builtin_amdgcn_wmma_f32_16x16x32_bf16(false, pa, false, v0,   (short)0, o0, false, false);
    o1 = __builtin_amdgcn_wmma_f32_16x16x32_bf16(false, pa, false, v1,   (short)0, o1, false, false);
    o2 = __builtin_amdgcn_wmma_f32_16x16x32_bf16(false, pa, false, v2,   (short)0, o2, false, false);
    o3 = __builtin_amdgcn_wmma_f32_16x16x32_bf16(false, pa, false, v3,   (short)0, o3, false, false);
    o4 = __builtin_amdgcn_wmma_f32_16x16x32_bf16(false, pa, false, ones, (short)0, o4, false, false);
  }

  // ---- epilogue: normalize by l (= o4) and store ctx bf16 [b][s][h*64+v] ----
#pragma unroll
  for (int r = 0; r < 8; ++r) {
    const int q = qt * 16 + r + 8 * hi;
    float inv = 1.0f / o4[r];
    unsigned short* crow = Ctx + ((size_t)(b * SEQ + q)) * EMB + h * DV;
    crow[m]      = f2bf(o0[r] * inv);
    crow[16 + m] = f2bf(o1[r] * inv);
    crow[32 + m] = f2bf(o2[r] * inv);
    crow[48 + m] = f2bf(o3[r] * inv);
  }
}

// ---------------- output projection + bias + residual ----------------
__global__ void __launch_bounds__(128) k_out(
    const unsigned short* __restrict__ Ctx,
    const unsigned short* __restrict__ WoutT,
    const float* __restrict__ bout,
    const float* __restrict__ x,
    float* __restrict__ out)
{
  const int lane = threadIdx.x & 31;
  const int wid  = blockIdx.x * 4 + (threadIdx.x >> 5);
  const int mt = wid / (EMB / 16), nt = wid % (EMB / 16);
  const int row0 = mt * 16, col0 = nt * 16;

  v8f acc = wave_gemm_bf16(Ctx + (size_t)row0 * EMB, EMB,
                           WoutT + (size_t)col0 * EMB, EMB, EMB, lane);

  const int hi = lane >> 4, n = lane & 15;
  const int ge = col0 + n;
  const float bias = bout[ge];
#pragma unroll
  for (int r = 0; r < 8; ++r) {
    int gm = row0 + r + 8 * hi;
    out[(size_t)gm * EMB + ge] = x[(size_t)gm * EMB + ge] + acc[r] + bias;
  }
}

// ---------------- host launcher ----------------
extern "C" void kernel_launch(void* const* d_in, const int* in_sizes, int n_in,
                              void* d_out, int out_size, void* d_ws, size_t ws_size,
                              hipStream_t stream)
{
  (void)in_sizes; (void)n_in; (void)out_size; (void)ws_size;
  const float*         x    = (const float*)d_in[0];
  const unsigned char* mask = (const unsigned char*)d_in[1];
  const float*         Wqkv = (const float*)d_in[2];
  const float*         bqkv = (const float*)d_in[3];
  const float*         Wout = (const float*)d_in[4];
  const float*         bout = (const float*)d_in[5];
  float*               out  = (float*)d_out;

  char* ws = (char*)d_ws;
  unsigned short* Xb    = (unsigned short*)(ws);              //  8 MB  (8192x512 bf16)
  unsigned short* WqkvT = (unsigned short*)(ws + 8388608);    //  1.5MB (1536x512 bf16)
  unsigned short* WoutT = (unsigned short*)(ws + 9961472);    //  0.5MB (512x512 bf16)
  unsigned short* Qw    = (unsigned short*)(ws + 10485760);   //  8 MB  [b][h][s][64]
  unsigned short* Kw    = (unsigned short*)(ws + 18874368);   //  8 MB  [b][h][s][64]
  unsigned short* Vtw   = (unsigned short*)(ws + 27262976);   //  8 MB  [b][h][64][s]
  unsigned short* Ctx   = (unsigned short*)(ws + 35651584);   //  8 MB  (8192x512 bf16)

  k_cvt_x    <<<(ROWS * EMB / 4) / 256, 256, 0, stream>>>(x, Xb);
  k_cvt_wqkvT<<<(EMB * QKVC) / 256,     256, 0, stream>>>(Wqkv, WqkvT);
  k_cvt_woutT<<<(EMB * EMB) / 256,      256, 0, stream>>>(Wout, WoutT);
  k_qkv      <<<(ROWS / 16) * (QKVC / 16) / 4, 128, 0, stream>>>(Xb, WqkvT, bqkv, Qw, Kw, Vtw);
  k_attn     <<<BATCH * NHEAD * (SEQ / 16) / 4, 128, ATTN_LDS, stream>>>(Qw, Kw, Vtw, mask, Ctx);
  k_out      <<<(ROWS / 16) * (EMB / 16) / 4,  128, 0, stream>>>(Ctx, WoutT, bout, x, out);
}
